// WindowAttention_17652315587218
// MI455X (gfx1250) — matrix operand order repaired and checked by reading, not compiled, not run on hardware
//
#include <hip/hip_runtime.h>
#include <hip/hip_bf16.h>

// ---------------------------------------------------------------------------
// Swin-V2 window attention for gfx1250 (MI455X), wave32 + v_wmma_f32_16x16x32_f16
// ---------------------------------------------------------------------------

typedef __attribute__((ext_vector_type(16))) _Float16 v16h;
typedef __attribute__((ext_vector_type(8)))  _Float16 v8h;
typedef __attribute__((ext_vector_type(8)))  float    v8f;

#define DIM    384
#define HEADS  12
#define HD     32
#define NTOK   343
#define NPAD   352            // 22 * 16
#define NWIN   64
#define MROWS  (NWIN * NTOK)  // 21952, divisible by 16
#define QKV3   (3 * DIM)      // 1152
#define TBL    2197           // 13^3 CPB table rows

// ---- WMMA fragment loaders (layouts per CDNA5 ISA 7.12.2) ------------------
// A (16x32 f16): lanes 0-15 => M=lane, e0..7 = K0..7,  e8..15 = K16..23
//                lanes16-31 => M=lane-16, e0..7 = K8..15, e8..15 = K24..31
static __device__ __forceinline__ v16h load_fragA(const _Float16* base, int row,
                                                  int ld, int lane) {
  const int l16 = (lane >> 4) & 1;
  const _Float16* p = base + (size_t)row * ld + l16 * 8;
  v8h lo = *(const v8h*)(p);        // K = l16*8 .. +7
  v8h hi = *(const v8h*)(p + 16);   // K = 16 + l16*8 .. +7
  v16h r;
#pragma unroll
  for (int e = 0; e < 8; ++e) { r[e] = lo[e]; r[8 + e] = hi[e]; }
  return r;
}

// B (32x16 f16): value B(K,n) read from memory laid out as mem[n*ld + K].
// lanes 0-15 => n=lane,    e0..7 = K0..7,  e8..15 = K16..23
// lanes16-31 => n=lane-16, e0..7 = K8..15, e8..15 = K24..31
static __device__ __forceinline__ v16h load_fragB(const _Float16* base, int col,
                                                  int ld, int lane) {
  const int l16 = (lane >> 4) & 1;
  const _Float16* p = base + (size_t)col * ld + l16 * 8;
  v8h lo = *(const v8h*)(p);
  v8h hi = *(const v8h*)(p + 16);
  v16h r;
#pragma unroll
  for (int e = 0; e < 8; ++e) { r[e] = lo[e]; r[8 + e] = hi[e]; }
  return r;
}

static __device__ __forceinline__ v8f wmma_f16(v16h a, v16h b, v8f c) {
  return __builtin_amdgcn_wmma_f32_16x16x32_f16(false, a, false, b, (short)0, c,
                                                false, false);
}

// ---- 1) fp32 -> fp16 convert ----------------------------------------------
__global__ void cvt_f32_f16(const float* __restrict__ src,
                            _Float16* __restrict__ dst, int n) {
  int i = blockIdx.x * 256 + threadIdx.x;
  if (i < n) dst[i] = (_Float16)src[i];
}

// ---- 2) CPB MLP: bias16[t][h] = 16*sigmoid(relu(tbl@W1^T+b1)@W2^T) --------
__global__ void cpb_kernel(const float* __restrict__ table,
                           const float* __restrict__ w1,
                           const float* __restrict__ b1,
                           const float* __restrict__ w2,
                           float* __restrict__ bias16) {
  int t = blockIdx.x * 256 + threadIdx.x;
  if (t >= TBL) return;
  float g0 = table[t * 3 + 0], g1 = table[t * 3 + 1], g2 = table[t * 3 + 2];
  float acc[HEADS];
#pragma unroll
  for (int h = 0; h < HEADS; ++h) acc[h] = 0.f;
  for (int j = 0; j < 512; ++j) {
    float hv = fmaf(w1[j * 3 + 2], g2,
                    fmaf(w1[j * 3 + 1], g1, fmaf(w1[j * 3], g0, b1[j])));
    hv = fmaxf(hv, 0.f);
#pragma unroll
    for (int h = 0; h < HEADS; ++h) acc[h] = fmaf(w2[h * 512 + j], hv, acc[h]);
  }
#pragma unroll
  for (int h = 0; h < HEADS; ++h)
    bias16[t * HEADS + h] = 16.f / (1.f + __expf(-acc[h]));
}

// ---- 3) QKV GEMM: qkvh = f16(x @ qkv_w^T + [q_bias,0,v_bias]) -------------
// One wave computes a 16x64 strip (4 N-tiles): A frag loaded once per K-step,
// reused across 4 WMMAs.
__global__ __launch_bounds__(256) void qkv_gemm(
    const _Float16* __restrict__ xh, const _Float16* __restrict__ wh,
    const float* __restrict__ q_bias, const float* __restrict__ v_bias,
    _Float16* __restrict__ qkvh) {
  const int lane = threadIdx.x & 31;
  const int wave = threadIdx.x >> 5;
  const int NG = QKV3 / 64;   // 18 strips
  const int Mt = MROWS / 16;  // 1372
  int tile = blockIdx.x * 8 + wave;
  if (tile >= Mt * NG) return;
  int tm = tile / NG, tg = tile % NG;
  int arow = tm * 16 + (lane & 15);
  int bcol = tg * 64 + (lane & 15);
  v8f a0 = {}, a1 = {}, a2 = {}, a3 = {};
#pragma unroll 2
  for (int k = 0; k < DIM; k += 32) {
    __builtin_prefetch(xh + (size_t)arow * DIM + k + 64, 0, 0);  // next A slice
    v16h a  = load_fragA(xh + k, arow, DIM, lane);
    v16h b0 = load_fragB(wh + k, bcol, DIM, lane);
    v16h b1 = load_fragB(wh + k, bcol + 16, DIM, lane);
    v16h b2 = load_fragB(wh + k, bcol + 32, DIM, lane);
    v16h b3 = load_fragB(wh + k, bcol + 48, DIM, lane);
    a0 = wmma_f16(a, b0, a0);
    a1 = wmma_f16(a, b1, a1);
    a2 = wmma_f16(a, b2, a2);
    a3 = wmma_f16(a, b3, a3);
  }
  v8f accs[4] = {a0, a1, a2, a3};
#pragma unroll
  for (int s = 0; s < 4; ++s) {
    int n = tg * 64 + s * 16 + (lane & 15);
    float bias =
        (n < DIM) ? q_bias[n] : ((n < 2 * DIM) ? 0.f : v_bias[n - 2 * DIM]);
#pragma unroll
    for (int r = 0; r < 8; ++r) {
      int m = tm * 16 + ((lane < 16) ? r : r + 8);
      qkvh[(size_t)m * QKV3 + n] = (_Float16)(accs[s][r] + bias);
    }
  }
}

// ---- 4) per-row L2 normalize q,k; split & pad to (b,h,NPAD,HD); vT --------
__global__ void norm_split_kernel(const _Float16* __restrict__ qkvh,
                                  _Float16* __restrict__ qh,
                                  _Float16* __restrict__ kh,
                                  _Float16* __restrict__ vT) {
  int t = blockIdx.x * 256 + threadIdx.x;
  const int total = NWIN * HEADS * NPAD;
  if (t >= total) return;
  int b = t / (HEADS * NPAD);
  int rem = t % (HEADS * NPAD);
  int h = rem / NPAD, n = rem % NPAD;
  size_t bh = (size_t)(b * HEADS + h);
  _Float16* qrow = qh + (bh * NPAD + n) * HD;
  _Float16* krow = kh + (bh * NPAD + n) * HD;
  _Float16* vcol = vT + bh * HD * NPAD + n;
  if (n < NTOK) {
    const _Float16* src = qkvh + ((size_t)(b * NTOK + n)) * QKV3 + h * HD;
    float q[HD], k[HD], qs = 0.f, ks = 0.f;
#pragma unroll
    for (int d = 0; d < HD; ++d) {
      q[d] = (float)src[d];
      k[d] = (float)src[DIM + d];
      qs = fmaf(q[d], q[d], qs);
      ks = fmaf(k[d], k[d], ks);
    }
    float qi = 1.f / fmaxf(sqrtf(qs), 1e-12f);
    float ki = 1.f / fmaxf(sqrtf(ks), 1e-12f);
#pragma unroll
    for (int d = 0; d < HD; ++d) {
      qrow[d] = (_Float16)(q[d] * qi);
      krow[d] = (_Float16)(k[d] * ki);
      vcol[(size_t)d * NPAD] = src[2 * DIM + d];
    }
  } else {  // zero padding rows/cols so WMMA tiles read clean zeros
#pragma unroll
    for (int d = 0; d < HD; ++d) {
      qrow[d] = (_Float16)0.f;
      krow[d] = (_Float16)0.f;
      vcol[(size_t)d * NPAD] = (_Float16)0.f;
    }
  }
}

// ---- 5) fused attention per (row-tile, head, window) ----------------------
__global__ __launch_bounds__(256) void attn_kernel(
    const _Float16* __restrict__ qh, const _Float16* __restrict__ kh,
    const _Float16* __restrict__ vT, const float* __restrict__ mask,
    const float* __restrict__ logit_scale, const int* __restrict__ rel_index,
    const float* __restrict__ bias16, _Float16* __restrict__ Oh) {
  __shared__ __align__(16) float    Sf[16 * NPAD];
  __shared__ __align__(16) _Float16 Ph[16 * NPAD];
  __shared__ float red[256];
  __shared__ float rowmax[16];
  __shared__ float rowsum[16];
  __shared__ float Ot[16 * HD];

  const int it = blockIdx.x;   // 0..21  row tiles
  const int h  = blockIdx.y;   // head
  const int b  = blockIdx.z;   // window
  const int t = threadIdx.x, lane = t & 31, wave = t >> 5;

  const float scale = __expf(fminf(logit_scale[h], 4.6051702f));  // exp(min(ls, log 100))
  const size_t bh = (size_t)(b * HEADS + h);
  const _Float16* qb = qh + bh * NPAD * HD;
  const _Float16* kb = kh + bh * NPAD * HD;
  const _Float16* vb = vT + bh * HD * NPAD;

  // S = (qn @ kn^T)*scale + cpb_bias + mask ; K = hd = 32 = single WMMA step
  v16h a = load_fragA(qb, it * 16 + (lane & 15), HD, lane);
  for (int jt = wave; jt < NPAD / 16; jt += 8) {
    v16h bk = load_fragB(kb, jt * 16 + (lane & 15), HD, lane);
    v8f c = {};
    c = wmma_f16(a, bk, c);
    int j = jt * 16 + (lane & 15);
#pragma unroll
    for (int r = 0; r < 8; ++r) {
      int il = (lane < 16) ? r : r + 8;
      int i = it * 16 + il;
      float val = -1e30f;
      if (i < NTOK && j < NTOK) {
        int idx = rel_index[i * NTOK + j];
        val = c[r] * scale + bias16[idx * HEADS + h] +
              mask[((size_t)b * NTOK + i) * NTOK + j];
      }
      Sf[il * NPAD + j] = val;
    }
  }
  __syncthreads();

  // softmax over 352 padded cols (pads are -1e30 -> exp = 0); 16 threads/row
  const int r = t >> 4, cc = t & 15;
  float m = -1e30f;
  for (int j = cc; j < NPAD; j += 16) m = fmaxf(m, Sf[r * NPAD + j]);
  red[t] = m;
  __syncthreads();
  if (cc == 0) {
    float mm = -1e30f;
    for (int q = 0; q < 16; ++q) mm = fmaxf(mm, red[r * 16 + q]);
    rowmax[r] = mm;
  }
  __syncthreads();
  float rm = rowmax[r], s = 0.f;
  for (int j = cc; j < NPAD; j += 16) {
    float e = __expf(Sf[r * NPAD + j] - rm);
    Sf[r * NPAD + j] = e;
    s += e;
  }
  red[t] = s;
  __syncthreads();
  if (cc == 0) {
    float ss = 0.f;
    for (int q = 0; q < 16; ++q) ss += red[r * 16 + q];
    rowsum[r] = ss;  // >= 1 (diagonal term), safe
  }
  __syncthreads();
  float inv = 1.f / rowsum[r];
  for (int j = cc; j < NPAD; j += 16)
    Ph[r * NPAD + j] = (_Float16)(Sf[r * NPAD + j] * inv);
  Ot[t] = 0.f;
  Ot[t + 256] = 0.f;
  __syncthreads();

  // O = P @ V : K split over waves (11 steps of 32), LDS reduce
  for (int kk = wave; kk < NPAD / 32; kk += 8) {
    v16h ap = load_fragA((const _Float16*)Ph + kk * 32, (lane & 15), NPAD, lane);
    v16h b0 = load_fragB(vb + kk * 32, (lane & 15), NPAD, lane);
    v16h b1 = load_fragB(vb + kk * 32, 16 + (lane & 15), NPAD, lane);
    v8f c0 = {}, c1 = {};
    c0 = wmma_f16(ap, b0, c0);
    c1 = wmma_f16(ap, b1, c1);
#pragma unroll
    for (int rr = 0; rr < 8; ++rr) {
      int il = (lane < 16) ? rr : rr + 8;
      atomicAdd(&Ot[il * HD + (lane & 15)], c0[rr]);
      atomicAdd(&Ot[il * HD + 16 + (lane & 15)], c1[rr]);
    }
  }
  __syncthreads();
  for (int idx = t; idx < 16 * HD; idx += 256) {
    int il = idx / HD, d = idx % HD;
    int i = it * 16 + il;
    if (i < NTOK)
      Oh[((size_t)(b * NTOK + i)) * DIM + h * HD + d] = (_Float16)Ot[idx];
  }
}

// ---- 6) Proj GEMM: out = Oh @ proj_w^T + proj_b (fp32 out) ----------------
__global__ __launch_bounds__(256) void proj_gemm(
    const _Float16* __restrict__ Oh, const _Float16* __restrict__ pwh,
    const float* __restrict__ pb, float* __restrict__ out) {
  const int lane = threadIdx.x & 31;
  const int wave = threadIdx.x >> 5;
  const int NG = DIM / 64;    // 6 strips
  const int Mt = MROWS / 16;  // 1372
  int tile = blockIdx.x * 8 + wave;
  if (tile >= Mt * NG) return;
  int tm = tile / NG, tg = tile % NG;
  int arow = tm * 16 + (lane & 15);
  int bcol = tg * 64 + (lane & 15);
  v8f a0 = {}, a1 = {}, a2 = {}, a3 = {};
#pragma unroll 2
  for (int k = 0; k < DIM; k += 32) {
    __builtin_prefetch(Oh + (size_t)arow * DIM + k + 64, 0, 0);
    v16h a  = load_fragA(Oh + k, arow, DIM, lane);
    v16h b0 = load_fragB(pwh + k, bcol, DIM, lane);
    v16h b1 = load_fragB(pwh + k, bcol + 16, DIM, lane);
    v16h b2 = load_fragB(pwh + k, bcol + 32, DIM, lane);
    v16h b3 = load_fragB(pwh + k, bcol + 48, DIM, lane);
    a0 = wmma_f16(a, b0, a0);
    a1 = wmma_f16(a, b1, a1);
    a2 = wmma_f16(a, b2, a2);
    a3 = wmma_f16(a, b3, a3);
  }
  v8f accs[4] = {a0, a1, a2, a3};
#pragma unroll
  for (int s = 0; s < 4; ++s) {
    int n = tg * 64 + s * 16 + (lane & 15);
    float bias = pb[n];
#pragma unroll
    for (int r = 0; r < 8; ++r) {
      int m = tm * 16 + ((lane < 16) ? r : r + 8);
      out[(size_t)m * DIM + n] = accs[s][r] + bias;
    }
  }
}

// ---------------------------------------------------------------------------
extern "C" void kernel_launch(void* const* d_in, const int* in_sizes, int n_in,
                              void* d_out, int out_size, void* d_ws,
                              size_t ws_size, hipStream_t stream) {
  const float* x           = (const float*)d_in[0];
  const float* mask        = (const float*)d_in[1];
  const float* qkv_w       = (const float*)d_in[2];
  const float* q_bias      = (const float*)d_in[3];
  const float* v_bias      = (const float*)d_in[4];
  const float* logit_scale = (const float*)d_in[5];
  const float* cpb_w1      = (const float*)d_in[6];
  const float* cpb_b1      = (const float*)d_in[7];
  const float* cpb_w2      = (const float*)d_in[8];
  const float* proj_w      = (const float*)d_in[9];
  const float* proj_b      = (const float*)d_in[10];
  const float* rel_table   = (const float*)d_in[11];
  const int*   rel_index   = (const int*)d_in[12];
  float* out = (float*)d_out;

  // workspace carve-out (~140 MB total)
  char* p = (char*)d_ws;
  auto carve = [&](size_t bytes) {
    char* r = p;
    p += (bytes + 255) & ~(size_t)255;
    return r;
  };
  _Float16* xh    = (_Float16*)carve(sizeof(_Float16) * (size_t)MROWS * DIM);
  _Float16* wh    = (_Float16*)carve(sizeof(_Float16) * (size_t)QKV3 * DIM);
  _Float16* pwh   = (_Float16*)carve(sizeof(_Float16) * (size_t)DIM * DIM);
  _Float16* qkvh  = (_Float16*)carve(sizeof(_Float16) * (size_t)MROWS * QKV3);
  _Float16* qh    = (_Float16*)carve(sizeof(_Float16) * (size_t)NWIN * HEADS * NPAD * HD);
  _Float16* kh    = (_Float16*)carve(sizeof(_Float16) * (size_t)NWIN * HEADS * NPAD * HD);
  _Float16* vT    = (_Float16*)carve(sizeof(_Float16) * (size_t)NWIN * HEADS * HD * NPAD);
  _Float16* Oh    = (_Float16*)carve(sizeof(_Float16) * (size_t)MROWS * DIM);
  float*    bias16 = (float*)carve(sizeof(float) * (size_t)TBL * HEADS);

  // 1) converts
  {
    int n = MROWS * DIM;
    cvt_f32_f16<<<(n + 255) / 256, 256, 0, stream>>>(x, xh, n);
    n = QKV3 * DIM;
    cvt_f32_f16<<<(n + 255) / 256, 256, 0, stream>>>(qkv_w, wh, n);
    n = DIM * DIM;
    cvt_f32_f16<<<(n + 255) / 256, 256, 0, stream>>>(proj_w, pwh, n);
  }
  // 2) CPB bias table
  cpb_kernel<<<(TBL + 255) / 256, 256, 0, stream>>>(rel_table, cpb_w1, cpb_b1,
                                                    cpb_w2, bias16);
  // 3) QKV GEMM  (1372 * 18 wave-strips, 8 waves/block)
  qkv_gemm<<<(MROWS / 16) * (QKV3 / 64) / 8, 256, 0, stream>>>(xh, wh, q_bias,
                                                               v_bias, qkvh);
  // 4) normalize + split + pad
  {
    int total = NWIN * HEADS * NPAD;
    norm_split_kernel<<<(total + 255) / 256, 256, 0, stream>>>(qkvh, qh, kh, vT);
  }
  // 5) fused attention
  attn_kernel<<<dim3(NPAD / 16, HEADS, NWIN), 256, 0, stream>>>(
      qh, kh, vT, mask, logit_scale, rel_index, bias16, Oh);
  // 6) projection GEMM (1372 * 6 wave-strips)
  proj_gemm<<<(MROWS / 16) * (DIM / 64) / 8, 256, 0, stream>>>(Oh, pwh, proj_b,
                                                               out);
  (void)in_sizes; (void)n_in; (void)out_size; (void)ws_size;
}
